// ConvTimeAware_LSTM_29386166239416
// MI455X (gfx1250) — matrix operand order repaired and checked by reading, not compiled, not run on hardware
//
#include <hip/hip_runtime.h>

typedef __attribute__((ext_vector_type(16))) _Float16 v16h;
typedef __attribute__((ext_vector_type(8)))  float    v8f;

#define B_   8
#define T_   16
#define CIN  3
#define HD   64
#define H_   64
#define W_   64
#define HW   (H_*W_)
#define CP   96          // packed input channels (x:3, h:64, t:1, pad:28)
#define TAPS 9

__device__ __forceinline__ float sigf(float x) { return 1.0f / (1.0f + __expf(-x)); }

// ---------------------------------------------------------------------------
// Repack fp32 weights [oc=64][Csrc][3][3] -> f16 [tap=9][oc=64][Kp] (zero pad)
// ---------------------------------------------------------------------------
__global__ void prep_weights(const float* __restrict__ src, int Csrc,
                             _Float16* __restrict__ dst, int Kp, int total) {
    for (int i = blockIdx.x * blockDim.x + threadIdx.x; i < total;
         i += gridDim.x * blockDim.x) {
        int k   = i % Kp;
        int oc  = (i / Kp) % HD;
        int tap = i / (Kp * HD);
        int kh = tap / 3, kw = tap % 3;
        float v = 0.0f;
        if (k < Csrc) v = src[((oc * Csrc + k) * 3 + kh) * 3 + kw];
        dst[i] = (_Float16)v;
    }
}

// ---------------------------------------------------------------------------
// Build NHWC f16 packed input P[b][y][x][96] = {x_t(3), h(64), tm(1), 0...}
// Optionally zero the f16 c-state buffer (t==0).
// ---------------------------------------------------------------------------
__global__ __launch_bounds__(256) void pack_input(
    const float* __restrict__ x,        // [B,T,CIN,H,W]
    const float* __restrict__ tm,       // [B,T,1,H,W]
    const _Float16* __restrict__ hprev, // NHWC [B,H,W,HD]
    _Float16* __restrict__ P,           // NHWC [B,H,W,CP]
    _Float16* __restrict__ c0,          // NHWC [B,H,W,HD]
    int t, int use_h, int zero_c)
{
    int pix = blockIdx.x * blockDim.x + threadIdx.x;   // b*HW + y*W + x
    if (pix >= B_ * HW) return;
    int b  = pix / HW;
    int yx = pix % HW;

    __align__(16) _Float16 v[CP];
#pragma unroll
    for (int c = 0; c < CIN; ++c)
        v[c] = (_Float16)x[((size_t)((b * T_ + t) * CIN + c)) * HW + yx];
    if (use_h) {
        const _Float16* hp = hprev + (size_t)pix * HD;
#pragma unroll
        for (int c = 0; c < HD; ++c) v[CIN + c] = hp[c];
    } else {
#pragma unroll
        for (int c = 0; c < HD; ++c) v[CIN + c] = (_Float16)0.0f;
    }
    v[CIN + HD] = (_Float16)tm[(size_t)(b * T_ + t) * HW + yx];
#pragma unroll
    for (int c = CIN + HD + 1; c < CP; ++c) v[c] = (_Float16)0.0f;

    uint4* dp = (uint4*)(P + (size_t)pix * CP);
    const uint4* sp = (const uint4*)v;
#pragma unroll
    for (int j = 0; j < CP / 8; ++j) dp[j] = sp[j];

    if (zero_c) {
        uint4 z; z.x = z.y = z.z = z.w = 0u;
        uint4* cp = (uint4*)(c0 + (size_t)pix * HD);
#pragma unroll
        for (int j = 0; j < HD / 8; ++j) cp[j] = z;
    }
}

// ---------------------------------------------------------------------------
// One LSTM timestep: implicit-GEMM 3x3 convs via WMMA + fused gate epilogue.
// Workgroup (8 waves) computes a 64-out-ch x 32-pixel tile (one b, one y).
// ---------------------------------------------------------------------------
__global__ __launch_bounds__(256) void lstm_step(
    const _Float16* __restrict__ P,     // NHWC [B,H,W,CP]
    const _Float16* __restrict__ Ccur,  // NHWC [B,H,W,HD]
    const _Float16* __restrict__ WAi,   // [9][64][96]
    const _Float16* __restrict__ WAct,  // [9][64][96]
    const _Float16* __restrict__ WAo,   // [9][64][96]
    const _Float16* __restrict__ WAcs,  // [9][64][64]
    const float* __restrict__ bi,  const float* __restrict__ bct,
    const float* __restrict__ bo,  const float* __restrict__ bcs,
    float* __restrict__ out,            // [B,T,HD,H,W]
    _Float16* __restrict__ Hnext,       // NHWC [B,H,W,HD]
    _Float16* __restrict__ Cnext,       // NHWC [B,H,W,HD]
    float* __restrict__ hlast, float* __restrict__ clast,
    int t, int write_last)
{
    __shared__ __align__(16) _Float16 Pls[3 * 34 * CP];  // rows y-1..y+1, x0-1..x0+32
    __shared__ __align__(16) _Float16 Cls[3 * 34 * HD];

    const int x0  = blockIdx.x * 32;
    const int y   = blockIdx.y;
    const int b   = blockIdx.z;
    const int tid = threadIdx.x;

    // ---- stage input halo slabs into LDS (zero-filled borders) ----
    {
        const int nchunkP = 3 * 34 * (CP / 8);   // 16B chunks
        for (int i = tid; i < nchunkP; i += 256) {
            int cpart = i % (CP / 8);
            int pixel = i / (CP / 8);
            int row = pixel / 34, xl = pixel % 34;
            int gy = y + row - 1, gx = x0 + xl - 1;
            uint4 val; val.x = val.y = val.z = val.w = 0u;
            if ((unsigned)gy < (unsigned)H_ && (unsigned)gx < (unsigned)W_)
                val = ((const uint4*)(P + ((size_t)((b * H_ + gy) * W_ + gx)) * CP))[cpart];
            ((uint4*)Pls)[i] = val;
        }
        const int nchunkC = 3 * 34 * (HD / 8);
        for (int i = tid; i < nchunkC; i += 256) {
            int cpart = i % (HD / 8);
            int pixel = i / (HD / 8);
            int row = pixel / 34, xl = pixel % 34;
            int gy = y + row - 1, gx = x0 + xl - 1;
            uint4 val; val.x = val.y = val.z = val.w = 0u;
            if ((unsigned)gy < (unsigned)H_ && (unsigned)gx < (unsigned)W_)
                val = ((const uint4*)(Ccur + ((size_t)((b * H_ + gy) * W_ + gx)) * HD))[cpart];
            ((uint4*)Cls)[i] = val;
        }
    }
    __syncthreads();

    const int lane = tid & 31;
    const int wid  = tid >> 5;
    const int m    = wid & 3;    // out-channel tile (16 rows)
    const int n    = wid >> 2;   // pixel tile (16 cols)
    const int l16  = lane & 15;
    const bool hi  = lane >= 16;

    v8f acc_i = {}, acc_ct = {}, acc_o = {}, acc_cs = {};

    const int npix  = 16 * n + l16;       // local pixel 0..31
    const int ocrow = 16 * m + l16;       // A-matrix row (out channel)

    for (int tap = 0; tap < TAPS; ++tap) {
        const int dy  = tap / 3 - 1, dx = tap % 3 - 1;
        const int row = dy + 1;
        const int xl  = npix + dx + 1;    // 0..33, halo handled by zero-fill
        const _Float16* pb = &Pls[(row * 34 + xl) * CP];
        const _Float16* pc = &Cls[(row * 34 + xl) * HD];

        // --- three shared-input convs over packed P (K = 96 = 3 x 32) ---
#pragma unroll
        for (int kc = 0; kc < 3; ++kc) {
            // B fragment: lane = N; K contiguous (0..15 lanes lo / 16..31 lanes hi)
            const int kb = kc * 32 + (hi ? 16 : 0);
            v16h bf;
            ((uint4*)&bf)[0] = *(const uint4*)(pb + kb);
            ((uint4*)&bf)[1] = *(const uint4*)(pb + kb + 8);

            // A fragment: lane = M; K contiguous (0..15 lanes lo / 16..31 lanes hi)
            const int ka = kc * 32 + (hi ? 16 : 0);
            const size_t wbase = ((size_t)(tap * HD + ocrow)) * CP + ka;

            v16h af;
            ((uint4*)&af)[0] = *(const uint4*)(WAi + wbase);
            ((uint4*)&af)[1] = *(const uint4*)(WAi + wbase + 8);
            acc_i = __builtin_amdgcn_wmma_f32_16x16x32_f16(
                false, af, false, bf, (short)0, acc_i, false, false);

            ((uint4*)&af)[0] = *(const uint4*)(WAct + wbase);
            ((uint4*)&af)[1] = *(const uint4*)(WAct + wbase + 8);
            acc_ct = __builtin_amdgcn_wmma_f32_16x16x32_f16(
                false, af, false, bf, (short)0, acc_ct, false, false);

            ((uint4*)&af)[0] = *(const uint4*)(WAo + wbase);
            ((uint4*)&af)[1] = *(const uint4*)(WAo + wbase + 8);
            acc_o = __builtin_amdgcn_wmma_f32_16x16x32_f16(
                false, af, false, bf, (short)0, acc_o, false, false);
        }

        // --- c-state conv (K = 64 = 2 x 32) ---
#pragma unroll
        for (int kc = 0; kc < 2; ++kc) {
            const int kb = kc * 32 + (hi ? 16 : 0);
            v16h bf;
            ((uint4*)&bf)[0] = *(const uint4*)(pc + kb);
            ((uint4*)&bf)[1] = *(const uint4*)(pc + kb + 8);

            const int ka = kc * 32 + (hi ? 16 : 0);
            const size_t wbase = ((size_t)(tap * HD + ocrow)) * HD + ka;
            v16h af;
            ((uint4*)&af)[0] = *(const uint4*)(WAcs + wbase);
            ((uint4*)&af)[1] = *(const uint4*)(WAcs + wbase + 8);
            acc_cs = __builtin_amdgcn_wmma_f32_16x16x32_f16(
                false, af, false, bf, (short)0, acc_cs, false, false);
        }
    }

    // ---- fused gate epilogue ----
    const int gx  = x0 + npix;
    const int ocb = 16 * m + (hi ? 8 : 0);   // C/D layout: 8 M rows per lane

    const _Float16* pcold = &Cls[(1 * 34 + (npix + 1)) * HD + ocb];
    const float tmv   = (float)Pls[(1 * 34 + (npix + 1)) * CP + (CIN + HD)];
    const float th_tm = tanhf(tmv);

    __align__(16) _Float16 hv[8];
    __align__(16) _Float16 cv[8];

#pragma unroll
    for (int r = 0; r < 8; ++r) {
        const int oc = ocb + r;
        const float iv     = sigf(acc_i[r] + bi[oc]);     // f_m == i_m (ref bug)
        const float cS     = acc_cs[r] + bcs[oc];
        const float cold   = (float)pcold[r];
        const float c_star = cold - cS * (1.0f + th_tm);  // (c - cS) - cS*tanh(tm)
        const float ctl    = tanhf(acc_ct[r] + bct[oc]);
        const float cnew   = iv * c_star + iv * ctl;
        const float ov     = sigf(acc_o[r] + bo[oc]);
        const float hnew   = ov * tanhf(cnew);

        out[((((size_t)b * T_ + t) * HD + oc) * H_ + y) * W_ + gx] = hnew;
        hv[r] = (_Float16)hnew;
        cv[r] = (_Float16)cnew;
        if (write_last) {
            hlast[(((size_t)b * HD + oc) * H_ + y) * W_ + gx] = hnew;
            clast[(((size_t)b * HD + oc) * H_ + y) * W_ + gx] = cnew;
        }
    }

    const size_t pixbase = ((size_t)((b * H_ + y) * W_ + gx)) * HD + ocb;
    *(uint4*)(Hnext + pixbase) = ((uint4*)hv)[0];
    *(uint4*)(Cnext + pixbase) = ((uint4*)cv)[0];
}

// ---------------------------------------------------------------------------
extern "C" void kernel_launch(void* const* d_in, const int* in_sizes, int n_in,
                              void* d_out, int out_size, void* d_ws, size_t ws_size,
                              hipStream_t stream) {
    const float* x   = (const float*)d_in[0];
    const float* tmt = (const float*)d_in[1];
    const float* Wi  = (const float*)d_in[2];
    const float* bi  = (const float*)d_in[3];
    const float* Wct = (const float*)d_in[4];
    const float* bct = (const float*)d_in[5];
    const float* Wo  = (const float*)d_in[6];
    const float* bo  = (const float*)d_in[7];
    const float* Wcs = (const float*)d_in[8];
    const float* bcs = (const float*)d_in[9];

    float* out   = (float*)d_out;                         // [B,T,HD,H,W]
    float* hlast = out + (size_t)B_ * T_ * HD * HW;       // [B,HD,H,W]
    float* clast = hlast + (size_t)B_ * HD * HW;          // [B,HD,H,W]

    char* ws = (char*)d_ws;
    size_t off = 0;
    auto alloc = [&](size_t bytes) -> void* {
        void* p = ws + off;
        off = (off + bytes + 255) & ~(size_t)255;
        return p;
    };
    _Float16* WAi  = (_Float16*)alloc((size_t)TAPS * HD * CP * 2);
    _Float16* WAct = (_Float16*)alloc((size_t)TAPS * HD * CP * 2);
    _Float16* WAo  = (_Float16*)alloc((size_t)TAPS * HD * CP * 2);
    _Float16* WAcs = (_Float16*)alloc((size_t)TAPS * HD * HD * 2);
    _Float16* Pbuf = (_Float16*)alloc((size_t)B_ * HW * CP * 2);
    _Float16* Hf   = (_Float16*)alloc((size_t)B_ * HW * HD * 2);
    _Float16* Cf0  = (_Float16*)alloc((size_t)B_ * HW * HD * 2);
    _Float16* Cf1  = (_Float16*)alloc((size_t)B_ * HW * HD * 2);
    _Float16* Cf[2] = {Cf0, Cf1};

    // Weight repack (tiny, once per call)
    prep_weights<<<dim3(216), 256, 0, stream>>>(Wi,  CIN + HD,     WAi,  CP, TAPS * HD * CP);
    prep_weights<<<dim3(216), 256, 0, stream>>>(Wct, CIN + HD,     WAct, CP, TAPS * HD * CP);
    prep_weights<<<dim3(216), 256, 0, stream>>>(Wo,  CIN + HD + 1, WAo,  CP, TAPS * HD * CP);
    prep_weights<<<dim3(144), 256, 0, stream>>>(Wcs, HD,           WAcs, HD, TAPS * HD * HD);

    const dim3 cgrid(W_ / 32, H_, B_);
    const int packBlocks = (B_ * HW + 255) / 256;
    int cur = 0;
    for (int t = 0; t < T_; ++t) {
        pack_input<<<packBlocks, 256, 0, stream>>>(
            x, tmt, Hf, Pbuf, Cf[0], t, /*use_h=*/(t > 0), /*zero_c=*/(t == 0));
        lstm_step<<<cgrid, 256, 0, stream>>>(
            Pbuf, Cf[cur], WAi, WAct, WAo, WAcs, bi, bct, bo, bcs,
            out, Hf, Cf[cur ^ 1], hlast, clast, t, /*write_last=*/(t == T_ - 1));
        cur ^= 1;
    }
}
